// WindowAttention_69166153335117
// MI455X (gfx1250) — matrix operand order repaired and checked by reading, not compiled, not run on hardware
//
#include <hip/hip_runtime.h>
#include <hip/hip_bf16.h>
#include <math.h>

typedef __bf16 bf16_t;
typedef __attribute__((ext_vector_type(16))) bf16_t v16bf;
typedef __attribute__((ext_vector_type(8)))  bf16_t v8bf;
typedef __attribute__((ext_vector_type(4)))  bf16_t v4bf;
typedef __attribute__((ext_vector_type(8)))  float  v8f;

#define N_TOK 49
#define NP    64
#define C_DIM 128
#define H_NUM 4
#define HD    32
#define NW    64
#define QKV_OUT 384
#define SCALE_F 0.17677669529663688f  // 1/sqrt(32)

// A-matrix fragment (16x32 bf16) from a row-major matrix (rows = A rows, contiguous K).
__device__ __forceinline__ v16bf load_frag_a(const bf16_t* base, int ld, int lane) {
  const int m  = lane & 15;
  const int kb = (lane >> 4) << 3;
  const bf16_t* p = base + m * ld + kb;
  v8bf lo = *(const v8bf*)(p);
  v8bf hi = *(const v8bf*)(p + 16);
  return __builtin_shufflevector(lo, hi, 0,1,2,3,4,5,6,7,8,9,10,11,12,13,14,15);
}

// B-matrix fragment (32x16 bf16) from B stored TRANSPOSED (N x K row-major).
__device__ __forceinline__ v16bf load_frag_b(const bf16_t* base, int ld, int lane) {
  const int n  = lane & 15;
  const int kb = (lane >> 4) << 3;
  const bf16_t* p = base + n * ld + kb;
  v8bf lo = *(const v8bf*)(p);
  v8bf hi = *(const v8bf*)(p + 16);
  return __builtin_shufflevector(lo, hi, 0,1,2,3,4,5,6,7,8,9,10,11,12,13,14,15);
}

__device__ __forceinline__ v8f wmma_bf16(v16bf a, v16bf b, v8f c) {
  return __builtin_amdgcn_wmma_f32_16x16x32_bf16(false, a, false, b, (short)0, c, false, false);
}

__device__ __forceinline__ v8bf cvt_v8(v8f a) {
  v8bf r;
#pragma unroll
  for (int j = 0; j < 8; ++j) r[j] = (bf16_t)a[j];
  return r;
}

// ---- prep: weights f32->bf16 (q-slice pre-scaled), q-bias pre-scaled ----
__global__ void prep_kernel(const float* __restrict__ wqkv,
                            const float* __restrict__ wproj,
                            const float* __restrict__ bqkv,
                            bf16_t* __restrict__ gw,
                            float*  __restrict__ gbq) {
  int i = blockIdx.x * blockDim.x + threadIdx.x;
  if (i < QKV_OUT * C_DIM) {
    float s = (i < C_DIM * C_DIM) ? SCALE_F : 1.f;       // rows 0..127 are q outputs
    gw[i] = (bf16_t)(wqkv[i] * s);
  }
  if (i < C_DIM * C_DIM) gw[QKV_OUT * C_DIM + i] = (bf16_t)wproj[i];
  if (i < QKV_OUT)       gbq[i] = bqkv[i] * ((i < C_DIM) ? SCALE_F : 1.f);
}

// ---- fused window attention: one block (8 waves) per window ----
__global__ void __launch_bounds__(256, 2)
swin_window_attn_kernel(const float* __restrict__ x,
                        const float* __restrict__ mask,
                        const float* __restrict__ rpb,
                        const int*   __restrict__ rel_idx,
                        const bf16_t* __restrict__ wqkv_bf,
                        const float* __restrict__ b_qkv_s,
                        const bf16_t* __restrict__ wproj_bf,
                        const float* __restrict__ b_proj,
                        float* __restrict__ out) {
  extern __shared__ char smem[];
  bf16_t* sX  = (bf16_t*)smem;                // 64x128 (16 KB); reused as attn-out
  bf16_t* sQ  = sX  + NP * C_DIM;             // [4][64][32] (token x d), q pre-scaled
  bf16_t* sK  = sQ  + H_NUM * NP * HD;        // [4][64][32] (token x d)
  bf16_t* sVt = sK  + H_NUM * NP * HD;        // [4][32][64] (d x token)
  bf16_t* sP  = sVt + H_NUM * HD * NP;        // [4][64][64] probs (query x key)

  const int b    = blockIdx.x;
  const int tid  = threadIdx.x;
  const int wave = tid >> 5;
  const int lane = tid & 31;
  const int hl   = lane >> 4;
  const int ln   = lane & 15;
  const int j8   = hl << 3;

  // ---- stage x tile as bf16 (vectorized), zero-pad rows 49..63 ----
  {
    const float4* xb4 = (const float4*)(x + (size_t)b * N_TOK * C_DIM);
#pragma unroll
    for (int it = 0; it < (NP * C_DIM / 4) / 256; ++it) {
      int i = it * 256 + tid;                 // float4 chunk index; row = i/32
      float4 v = {0.f, 0.f, 0.f, 0.f};
      if ((i >> 5) < N_TOK) v = xb4[i];
      v4bf o = {(bf16_t)v.x, (bf16_t)v.y, (bf16_t)v.z, (bf16_t)v.w};
      *(v4bf*)(sX + 4 * i) = o;
    }
  }
  __syncthreads();

  // ---- Q/K GEMM, transposed: D = Wqk(256x128) @ X^T(128x64) ----
  // lane = token column; j = 8 consecutive output channels -> packed 16B stores
  {
    const int tt = wave & 3;                   // token tile
    const int cb = (wave >> 2) << 3;           // channel-tile base (0 or 8)
    v16bf xb[4];
#pragma unroll
    for (int ks = 0; ks < 4; ++ks)
      xb[ks] = load_frag_b(sX + tt * 16 * C_DIM + ks * 32, C_DIM, lane);
    const int token = tt * 16 + ln;
#pragma unroll
    for (int i = 0; i < 8; ++i) {
      const int ct = cb + i;
      const int ob = ct * 16 + j8;             // this lane's first output channel
      v8f acc = *(const v8f*)(b_qkv_s + ob);   // vector bias init (q pre-scaled)
#pragma unroll
      for (int ks = 0; ks < 4; ++ks) {
        v16bf af = load_frag_a(wqkv_bf + ct * 16 * C_DIM + ks * 32, C_DIM, lane);
        acc = wmma_bf16(af, xb[ks], acc);
      }
      const int s  = ob >> 7;                  // 0=q, 1=k (uniform per wave)
      const int hh = (ob & 127) >> 5;
      const int db = ob & 31;
      *(v8bf*)(sQ + s * (H_NUM * NP * HD) + hh * NP * HD + token * HD + db) = cvt_v8(acc);
    }
  }
  // ---- V GEMM, normal: X(64x128) @ Wv^T(128x128); j = 8 consecutive tokens ----
  {
    v16bf wb[4];
#pragma unroll
    for (int ks = 0; ks < 4; ++ks)
      wb[ks] = load_frag_b(wqkv_bf + (2 * C_DIM + wave * 16) * C_DIM + ks * 32, C_DIM, lane);
    const int oc = 2 * C_DIM + wave * 16 + ln; // absolute v output channel
    const int hh = (oc & 127) >> 5;
    const int d  = oc & 31;
    const float bias = b_qkv_s[oc];
#pragma unroll
    for (int mt = 0; mt < 4; ++mt) {
      v8f acc = {bias, bias, bias, bias, bias, bias, bias, bias};
#pragma unroll
      for (int ks = 0; ks < 4; ++ks) {
        v16bf af = load_frag_a(sX + mt * 16 * C_DIM + ks * 32, C_DIM, lane);
        acc = wmma_bf16(af, wb[ks], acc);
      }
      *(v8bf*)(sVt + hh * HD * NP + d * NP + mt * 16 + j8) = cvt_v8(acc);
    }
  }
  __syncthreads();

  // ---- attention: 2 waves per head; scores transposed: S^T = K @ Q^T ----
  const int h  = wave >> 1;
  const int mh = wave & 1;
  const bf16_t* sQh = sQ  + h * NP * HD;
  const bf16_t* sKh = sK  + h * NP * HD;
  const bf16_t* sVh = sVt + h * HD * NP;
  bf16_t*       sPh = sP  + h * NP * NP;
  const int wofs = (b & (NW - 1)) * N_TOK * N_TOK;

#pragma unroll
  for (int qi = 0; qi < 2; ++qi) {
    const int qt = mh * 2 + qi;
    const int q  = qt * 16 + ln;               // this lane's query token
    v16bf bq = load_frag_b(sQh + qt * 16 * HD, HD, lane);   // B = Q^T, K = hd = 32
    v8f sc[4];
#pragma unroll
    for (int kt = 0; kt < 4; ++kt) {
      v16bf ak = load_frag_a(sKh + kt * 16 * HD, HD, lane); // A = K tokens
      v8f z = {};
      sc[kt] = wmma_bf16(ak, bq, z);
    }
    // bias + mask (branchless clamped gather); lane owns one query row
    const int qc   = (q < N_TOK) ? q : (N_TOK - 1);
    const int prow = qc * N_TOK;
#pragma unroll
    for (int kt = 0; kt < 4; ++kt) {
#pragma unroll
      for (int j = 0; j < 8; ++j) {
        const int mk  = kt * 16 + j8 + j;
        const int mkc = (mk < N_TOK) ? mk : (N_TOK - 1);
        const float add = rpb[rel_idx[prow + mkc] * H_NUM + h] + mask[wofs + prow + mkc];
        float v = sc[kt][j];
        v = (q < N_TOK && mk < N_TOK) ? (v + add) : v;
        v = (mk >= N_TOK) ? -__builtin_inff() : v;
        sc[kt][j] = v;
      }
    }
    // softmax: this lane + partner lane (^16) hold the full 64-key row
    float mx = -__builtin_inff();
#pragma unroll
    for (int kt = 0; kt < 4; ++kt)
#pragma unroll
      for (int j = 0; j < 8; ++j) mx = fmaxf(mx, sc[kt][j]);
    mx = fmaxf(mx, __shfl_xor(mx, 16, 32));
    float sum = 0.f;
#pragma unroll
    for (int kt = 0; kt < 4; ++kt)
#pragma unroll
      for (int j = 0; j < 8; ++j) { float e = __expf(sc[kt][j] - mx); sc[kt][j] = e; sum += e; }
    sum += __shfl_xor(sum, 16, 32);
    const float inv = 1.f / sum;
#pragma unroll
    for (int kt = 0; kt < 4; ++kt)
      *(v8bf*)(sPh + q * NP + kt * 16 + j8) = cvt_v8(sc[kt] * inv);  // padded keys -> 0

    // attn @ V, transposed: O^T = V^T(32x64) @ P^T(64x16); j = 8 consecutive channels
    v16bf bp[2];
#pragma unroll
    for (int ks = 0; ks < 2; ++ks)
      bp[ks] = load_frag_b(sPh + qt * 16 * NP + ks * 32, NP, lane);
#pragma unroll
    for (int dt = 0; dt < 2; ++dt) {
      v8f acc = {};
#pragma unroll
      for (int ks = 0; ks < 2; ++ks) {
        v16bf av = load_frag_a(sVh + dt * 16 * NP + ks * 32, NP, lane);
        acc = wmma_bf16(av, bp[ks], acc);
      }
      *(v8bf*)(sX + q * C_DIM + h * HD + dt * 16 + j8) = cvt_v8(acc);
    }
  }
  __syncthreads();

  // ---- projection, transposed: D = Wp(128x128) @ O^T(128x64) ----
  {
    const int tt = wave & 3;
    const int cb = (wave >> 2) << 2;           // channel-tile base (0 or 4)
    v16bf xo[4];
#pragma unroll
    for (int ks = 0; ks < 4; ++ks)
      xo[ks] = load_frag_b(sX + tt * 16 * C_DIM + ks * 32, C_DIM, lane);
    const int token = tt * 16 + ln;
    float* ob = out + (size_t)b * N_TOK * C_DIM + token * C_DIM;
#pragma unroll
    for (int i = 0; i < 4; ++i) {
      const int ct = cb + i;
      const int cbase = ct * 16 + j8;          // 8 consecutive out channels
      v8f acc = *(const v8f*)(b_proj + cbase); // vector bias init
#pragma unroll
      for (int ks = 0; ks < 4; ++ks) {
        v16bf af = load_frag_a(wproj_bf + ct * 16 * C_DIM + ks * 32, C_DIM, lane);
        acc = wmma_bf16(af, xo[ks], acc);
      }
      if (token < N_TOK) *(v8f*)(ob + cbase) = acc;   // 32B contiguous f32 store
    }
  }
}

extern "C" void kernel_launch(void* const* d_in, const int* in_sizes, int n_in,
                              void* d_out, int out_size, void* d_ws, size_t ws_size,
                              hipStream_t stream) {
  (void)in_sizes; (void)n_in; (void)out_size; (void)ws_size;
  const float* x     = (const float*)d_in[0];
  const float* mask  = (const float*)d_in[1];
  const float* rpb   = (const float*)d_in[2];
  const int*   reli  = (const int*)d_in[3];
  const float* wqkv  = (const float*)d_in[4];
  const float* bqkv  = (const float*)d_in[5];
  const float* wproj = (const float*)d_in[6];
  const float* bproj = (const float*)d_in[7];
  float* out = (float*)d_out;

  // workspace: [wqkv bf16 49152][wproj bf16 16384][b_qkv_scaled f32 384]
  bf16_t* wbf = (bf16_t*)d_ws;
  float*  bqs = (float*)((char*)d_ws + (QKV_OUT * C_DIM + C_DIM * C_DIM) * sizeof(bf16_t));
  prep_kernel<<<(QKV_OUT * C_DIM + 255) / 256, 256, 0, stream>>>(wqkv, wproj, bqkv, wbf, bqs);

  const size_t smem =
      (size_t)(NP * C_DIM + 3 * H_NUM * NP * HD + H_NUM * NP * NP) * sizeof(bf16_t); // 96 KB
  hipFuncSetAttribute(reinterpret_cast<const void*>(swin_window_attn_kernel),
                      hipFuncAttributeMaxDynamicSharedMemorySize, (int)smem);
  swin_window_attn_kernel<<<4096, 256, smem, stream>>>(
      x, mask, rpb, reli, wbf, bqs, wbf + QKV_OUT * C_DIM, bproj, out);
}
